// spectralModel_66967130079324
// MI455X (gfx1250) — matrix operand order repaired and checked by reading, not compiled, not run on hardware
//
#include <hip/hip_runtime.h>
#include <math.h>

// ---------------- model constants ----------------
constexpr int   kB    = 2;
constexpr int   kCin  = 3;
constexpr int   kC    = 128;
constexpr int   kNH   = 8;
constexpr int   kHD   = 16;
constexpr int   kH    = 64;
constexpr int   kW    = 64;
constexpr int   kHF   = 64;
constexpr int   kWF   = 33;            // rfft bins along W
constexpr int   kHW   = kHF * kWF;     // 2112 spectral tokens (132 tiles of 16)
constexpr float kEPS  = 1.1920929e-07f;
constexpr float kSCL  = 0.25f;         // 1/sqrt(HD)

// ---------------- CDNA5 WMMA helpers (wave32) ----------------
typedef __attribute__((ext_vector_type(16))) _Float16 v16h;
typedef __attribute__((ext_vector_type(8)))  _Float16 v8h;
typedef __attribute__((ext_vector_type(8)))  float    v8f;

__device__ __forceinline__ v8f wmma16(v16h a, v16h b, v8f c) {
  // D = A(16x32 f16) * B(32x16 f16) + C(16x16 f32)
  return __builtin_amdgcn_wmma_f32_16x16x32_f16(
      /*neg_a=*/false, a, /*neg_b=*/false, b,
      /*c_mod=*/(short)0, c, /*reuse_a=*/false, /*reuse_b=*/false);
}

// ISA 7.12.2: 16-bit A 16x32 layout. lane<16: row=lane, K={0..7,16..23};
// lane>=16: row=lane-16, K={8..15,24..31}. j-th packed f16 -> K index:
__device__ __forceinline__ int a_kidx(int lane, int j) {
  int kbase = (lane < 16) ? 0 : 8;
  return kbase + j + ((j >= 8) ? 8 : 0);
}

union V16U { v16h v; v8h h[2]; };

// ---------------- DFT twiddle tables (ortho-scaled, 1/8 per dim) ----------------
__global__ void dft_init_kernel(float* __restrict__ Dre, float* __restrict__ Dim) {
  int t = blockIdx.x * blockDim.x + threadIdx.x;
  if (t < 64 * 64) {
    int k = t >> 6, n = t & 63;
    float th = 6.28318530717958647692f * (float)((k * n) & 63) / 64.0f;
    float s, c;
    __sincosf(th, &s, &c);
    Dre[t] = 0.125f * c;
    Dim[t] = -0.125f * s;
  }
}

// rfft2 (ortho) of one 64x64 real image per block -> 64x33 complex tokens.
__global__ void __launch_bounds__(256)
rfft2_kernel(const float* __restrict__ X, float2* __restrict__ F,
             const float* __restrict__ Dre, const float* __restrict__ Dim) {
  __shared__ float  sx[kH * kW];
  __shared__ float2 mid[kH * kWF];
  const float* in  = X + (size_t)blockIdx.x * kH * kW;
  float2*      out = F + (size_t)blockIdx.x * kHW;
  for (int i = threadIdx.x; i < kH * kW; i += blockDim.x) sx[i] = in[i];
  __syncthreads();
  for (int i = threadIdx.x; i < kH * kWF; i += blockDim.x) {   // along W
    int h = i / kWF, kw = i % kWF;
    float re = 0.f, im = 0.f;
    for (int w = 0; w < kW; ++w) {
      float v = sx[h * kW + w];
      re += v * Dre[kw * 64 + w];
      im += v * Dim[kw * 64 + w];
    }
    mid[i] = make_float2(re, im);
  }
  __syncthreads();
  for (int i = threadIdx.x; i < kHF * kWF; i += blockDim.x) {  // along H
    int kh = i / kWF, kw = i % kWF;
    float re = 0.f, im = 0.f;
    for (int h = 0; h < kH; ++h) {
      float cr = Dre[kh * 64 + h], ci = Dim[kh * 64 + h];
      float2 m = mid[h * kWF + kw];
      re += cr * m.x - ci * m.y;
      im += cr * m.y + ci * m.x;
    }
    out[i] = make_float2(re, im);
  }
}

// irfft2 (ortho): 64x33 complex tokens per block -> 64x64 real image.
__global__ void __launch_bounds__(256)
irfft2_kernel(const float2* __restrict__ F, float* __restrict__ X,
              const float* __restrict__ Dre, const float* __restrict__ Dim) {
  __shared__ float2 sf[kHF * kWF];
  __shared__ float2 mid[kH * kWF];
  const float2* in  = F + (size_t)blockIdx.x * kHW;
  float*        out = X + (size_t)blockIdx.x * kH * kW;
  for (int i = threadIdx.x; i < kHF * kWF; i += blockDim.x) sf[i] = in[i];
  __syncthreads();
  for (int i = threadIdx.x; i < kH * kWF; i += blockDim.x) {   // inverse along H
    int h = i / kWF, kw = i % kWF;
    float re = 0.f, im = 0.f;
    for (int kh = 0; kh < kHF; ++kh) {
      float cr = Dre[kh * 64 + h], ci = -Dim[kh * 64 + h];     // conj = e^{+i..}/8
      float2 m = sf[kh * kWF + kw];
      re += cr * m.x - ci * m.y;
      im += cr * m.y + ci * m.x;
    }
    mid[i] = make_float2(re, im);
  }
  __syncthreads();
  for (int i = threadIdx.x; i < kH * kW; i += blockDim.x) {    // real inverse along W
    int h = i / kW, w = i % kW;
    float acc = 0.f;
    for (int kw = 0; kw < kWF; ++kw) {
      float wk = (kw == 0 || kw == 32) ? 1.f : 2.f;            // Hermitian fold
      float cr = Dre[kw * 64 + w], ci = -Dim[kw * 64 + w];
      float2 m = mid[h * kWF + kw];
      acc += wk * (m.x * cr - m.y * ci);
    }
    out[i] = acc;
  }
}

// Complex channel GEMM: y[b,o,t] = sum_i f[b,i,t] * Wm[i*C+o]  (+residual).
// All 8 waves of a block share one o-tile: the 16x128 complex weight slice is
// staged ONCE per block into LDS via async global->LDS copies (ASYNCcnt),
// converted to padded f16 planes, and each WMMA A-fragment becomes two
// 16B ds_load_b128 reads instead of 16 scattered global loads.
__global__ void __launch_bounds__(256)
cgemm128_kernel(const float2* __restrict__ F, const float2* __restrict__ Wm,
                const float2* __restrict__ Res, float2* __restrict__ Out) {
  constexpr int kTT  = kHW / 16;         // 132 t-tiles
  constexpr int kTG  = (kTT + 7) / 8;    // 17 groups of 8 tiles
  constexpr int kPAD = 136;              // f16 row stride (bank rotation)
  __shared__ float2   sW[kC * 16];       // raw staged weights (16KB)
  __shared__ _Float16 sAre[16][kPAD];    // A planes, [o][i]
  __shared__ _Float16 sAim[16][kPAD];

  int tid  = threadIdx.x;
  int wid  = tid >> 5;
  int lane = tid & 31;
  int bi   = blockIdx.x;
  int b    = bi / (8 * kTG);
  int rem  = bi % (8 * kTG);
  int o0   = (rem / kTG) * 16;
  int tg   = rem % kTG;

  // ---- async stage: 128 rows x 16 float2 = 1024 x 16B chunks --------------
  unsigned ldsW = (unsigned)(size_t)(void*)sW;
#pragma unroll
  for (int it = 0; it < 4; ++it) {
    int c = tid + it * 256;                              // chunk id 0..1023
    int i = c >> 3, seg = c & 7;
    unsigned long long ga =
        (unsigned long long)(const void*)(Wm + (size_t)i * kC + o0 + seg * 2);
    unsigned la = ldsW + (unsigned)c * 16u;
    asm volatile("global_load_async_to_lds_b128 %0, %1, off"
                 :: "v"(la), "v"(ga) : "memory");
  }
  asm volatile("s_wait_asynccnt 0" ::: "memory");
  __syncthreads();
  // convert to transposed f16 planes: sAre[o][i] = Wm[i][o0+o]
  for (int e = tid; e < kC * 16; e += 256) {
    int i = e >> 4, o = e & 15;
    float2 w = sW[e];
    sAre[o][i] = (_Float16)w.x;
    sAim[o][i] = (_Float16)w.y;
  }
  __syncthreads();

  // ---- per-wave 16x16 tile ------------------------------------------------
  int tt = tg * 8 + wid;
  if (tt < kTT) {
    int t0    = tt * 16;
    int am    = lane & 15;
    int nn    = lane & 15;
    int kbase = (lane < 16) ? 0 : 8;
    int kb2   = (lane < 16) ? 0 : 16;
    v8f arr = {}, aii = {}, ari = {}, air = {};
#pragma unroll
    for (int kc = 0; kc < kC; kc += 32) {
      // A fragments from LDS: two contiguous 8-half runs per plane
      V16U ua, ub;
      ua.h[0] = *(const v8h*)&sAre[am][kc + kbase];
      ua.h[1] = *(const v8h*)&sAre[am][kc + kbase + 16];
      ub.h[0] = *(const v8h*)&sAim[am][kc + kbase];
      ub.h[1] = *(const v8h*)&sAim[am][kc + kbase + 16];
      // warm L2 for the next K-chunk of B
      if (kc + 32 < kC)
        __builtin_prefetch(&F[((size_t)b * kC + kc + 32 + kb2) * kHW + t0 + nn], 0, 0);
      v16h Br, Bi;
#pragma unroll
      for (int p = 0; p < 8; ++p) {
        int k0 = kbase + 2 * p + ((p >= 4) ? 8 : 0);
        float2 f0 = F[((size_t)b * kC + kc + k0    ) * kHW + t0 + nn];
        float2 f1 = F[((size_t)b * kC + kc + k0 + 1) * kHW + t0 + nn];
        Br[2 * p]     = (_Float16)f0.x;  Br[2 * p + 1] = (_Float16)f1.x;
        Bi[2 * p]     = (_Float16)f0.y;  Bi[2 * p + 1] = (_Float16)f1.y;
      }
      arr = wmma16(ua.v, Br, arr);   // Wr*fr
      aii = wmma16(ub.v, Bi, aii);   // Wi*fi
      ari = wmma16(ua.v, Bi, ari);   // Wr*fi
      air = wmma16(ub.v, Br, air);   // Wi*fr
    }
    int mm = (lane < 16) ? 0 : 8;
#pragma unroll
    for (int r = 0; r < 8; ++r) {
      size_t oidx = ((size_t)b * kC + o0 + r + mm) * kHW + t0 + nn;
      float yr = arr[r] - aii[r];
      float yi = ari[r] + air[r];
      if (Res) { float2 rv = Res[oidx]; yr += rv.x; yi += rv.y; }
      Out[oidx] = make_float2(yr, yi);
    }
  }
}

// Fused spectral attention: 8 waves = 8 heads per block; head-axis softmax via
// LDS; causal mask post-softmax; never materializes the hw x hw score matrix.
__global__ void __launch_bounds__(256)
attn_kernel(const float2* __restrict__ Q, const float2* __restrict__ K,
            const float2* __restrict__ V, float2* __restrict__ O) {
  __shared__ float sS[kNH * 256];    // |score| per head, C/D element layout
  __shared__ float sP[kNH * 256];    // softmaxed p per head, (m,n) layout
  const int xtiles = kHW / 16;       // 132
  int xt   = blockIdx.x % xtiles;
  int b    = blockIdx.x / xtiles;
  int head = threadIdx.x >> 5;
  int lane = threadIdx.x & 31;
  int x0   = xt * 16;
  int am   = lane & 15;
  int mm   = (lane < 16) ? 0 : 8;
  int nn   = lane & 15;
  int kb2  = (lane < 16) ? 0 : 16;
  const float2* qb = Q + ((size_t)b * kC + head * kHD) * kHW;
  const float2* kb = K + ((size_t)b * kC + head * kHD) * kHW;
  const float2* vb = V + ((size_t)b * kC + head * kHD) * kHW;

  // A = [qr | qi] (16x32): rows = 16 queries, K packs re|im of the 16 dims.
  v16h Aq;
#pragma unroll
  for (int j = 0; j < 16; ++j) {
    int kk = a_kidx(lane, j);
    int d  = (kk < 16) ? kk : kk - 16;
    float2 qv = qb[(size_t)d * kHW + x0 + am];
    Aq[j] = (_Float16)((kk < 16) ? qv.x : qv.y);
  }
  v8f oR = {}, oI = {};
  for (int yt = 0; yt <= xt; ++yt) {           // causal: only y-tiles <= x-tile
    int y0 = yt * 16;
    if (yt < xt) {                             // warm L2 for next y-tile
      __builtin_prefetch(&kb[(size_t)am * kHW + y0 + 16], 0, 0);
      __builtin_prefetch(&vb[(size_t)am * kHW + y0 + 16], 0, 0);
    }
    // conj(q).k : re = qr*kr + qi*ki ; im = qr*ki - qi*kr
    v16h BrS, BiS;
#pragma unroll
    for (int p = 0; p < 8; ++p) {
#pragma unroll
      for (int q2 = 0; q2 < 2; ++q2) {
        int kk = a_kidx(lane, 2 * p + q2);
        int d  = (kk < 16) ? kk : kk - 16;
        float2 kv = kb[(size_t)d * kHW + y0 + nn];
        BrS[2 * p + q2] = (_Float16)((kk < 16) ? kv.x : kv.y);   // [kr ; ki]
        BiS[2 * p + q2] = (_Float16)((kk < 16) ? kv.y : -kv.x);  // [ki ; -kr]
      }
    }
    v8f zero = {};
    v8f sre = wmma16(Aq, BrS, zero);
    v8f sim = wmma16(Aq, BiS, zero);
#pragma unroll
    for (int r = 0; r < 8; ++r)
      sS[head * 256 + r * 32 + lane] =
          sqrtf(sre[r] * sre[r] + sim[r] * sim[r]) * kSCL;
    __syncthreads();
    // softmax across the 8 heads (reference: softmax over axis=1), then mask
#pragma unroll
    for (int r = 0; r < 8; ++r) {
      int e = r * 32 + lane;
      float mx = -3.0e38f;
      for (int h = 0; h < kNH; ++h) mx = fmaxf(mx, sS[h * 256 + e]);
      float den = 0.f;
      for (int h = 0; h < kNH; ++h) den += __expf(sS[h * 256 + e] - mx);
      float pv = __expf(sS[head * 256 + e] - mx) / den;
      int xg = x0 + r + mm, yg = y0 + nn;
      if (yg > xg) pv = 0.f;                    // triu mask, applied after softmax
      sP[head * 256 + (r + mm) * 16 + nn] = pv;
    }
    __syncthreads();
    // o += p * v  (A = p 16x16 zero-padded to K=32, B = [vr],[vi])
    v16h Ap, BvR, BvI;
#pragma unroll
    for (int j = 0; j < 16; ++j) {
      int kk = a_kidx(lane, j);
      Ap[j] = (kk < 16) ? (_Float16)sP[head * 256 + am * 16 + kk] : (_Float16)0.f;
    }
#pragma unroll
    for (int p = 0; p < 8; ++p) {
#pragma unroll
      for (int q2 = 0; q2 < 2; ++q2) {
        int kk = a_kidx(lane, 2 * p + q2);
        float2 vv = (kk < 16) ? vb[(size_t)nn * kHW + y0 + kk]
                              : make_float2(0.f, 0.f);
        BvR[2 * p + q2] = (_Float16)vv.x;
        BvI[2 * p + q2] = (_Float16)vv.y;
      }
    }
    oR = wmma16(Ap, BvR, oR);
    oI = wmma16(Ap, BvI, oI);
  }
#pragma unroll
  for (int r = 0; r < 8; ++r)
    O[((size_t)b * kC + head * kHD + nn) * kHW + x0 + r + mm] =
        make_float2(oR[r], oI[r]);
}

// RMSNorm over channel dim at each (b,h,w).
__global__ void rmsnorm_kernel(const float* __restrict__ X,
                               const float* __restrict__ Wt,
                               float* __restrict__ Y) {
  int idx = blockIdx.x * blockDim.x + threadIdx.x;
  if (idx >= kB * kH * kW) return;
  int b = idx / (kH * kW), p = idx % (kH * kW);
  const float* xb = X + (size_t)b * kC * kH * kW + p;
  float ss = 0.f;
  for (int c = 0; c < kC; ++c) { float v = xb[(size_t)c * kH * kW]; ss += v * v; }
  float inv = rsqrtf(ss / (float)kC + kEPS);
  float* yb = Y + (size_t)b * kC * kH * kW + p;
  for (int c = 0; c < kC; ++c)
    yb[(size_t)c * kH * kW] = xb[(size_t)c * kH * kW] * inv * Wt[c];
}

// Stem projection: 3 -> 128 complex channels (+bias). K=3, VALU is right.
__global__ void win_proj_kernel(const float2* __restrict__ F3,
                                const float2* __restrict__ Win,
                                const float2* __restrict__ Bin,
                                float2* __restrict__ Out) {
  int idx = blockIdx.x * blockDim.x + threadIdx.x;
  if (idx >= kB * kC * kHW) return;
  int t = idx % kHW, o = (idx / kHW) % kC, b = idx / (kHW * kC);
  float2 acc = Bin[o];
  for (int i = 0; i < kCin; ++i) {
    float2 w = Win[o * kCin + i];
    float2 f = F3[((size_t)b * kCin + i) * kHW + t];
    acc.x += w.x * f.x - w.y * f.y;
    acc.y += w.x * f.y + w.y * f.x;
  }
  Out[idx] = acc;
}

// Head projection: 128 -> 3 complex channels (+bias). M=3, VALU is right.
__global__ void wout_proj_kernel(const float2* __restrict__ F,
                                 const float2* __restrict__ Wout,
                                 const float2* __restrict__ Bout,
                                 float2* __restrict__ Out) {
  int idx = blockIdx.x * blockDim.x + threadIdx.x;
  if (idx >= kB * kCin * kHW) return;
  int t = idx % kHW, o = (idx / kHW) % kCin, b = idx / (kHW * kCin);
  float2 acc = Bout[o];
  for (int i = 0; i < kC; ++i) {
    float2 w = Wout[o * kC + i];
    float2 f = F[((size_t)b * kC + i) * kHW + t];
    acc.x += w.x * f.x - w.y * f.y;
    acc.y += w.x * f.y + w.y * f.x;
  }
  Out[idx] = acc;
}

__global__ void silu_add_kernel(const float* __restrict__ U,
                                const float* __restrict__ Xr,
                                float* __restrict__ Y, int n) {
  int i = blockIdx.x * blockDim.x + threadIdx.x;
  if (i < n) { float v = U[i]; Y[i] = v / (1.f + __expf(-v)) + Xr[i]; }
}

// d_out = [ out real (B*3*64*64) | xf complex64 interleaved (B*3*64*33*2) ]
__global__ void pack_kernel(const float* __restrict__ out3,
                            const float2* __restrict__ f3o,
                            float* __restrict__ dout) {
  const int n1 = kB * kCin * kH * kW;       // 24576
  const int n2 = 2 * kB * kCin * kHW;       // 25344 floats (interleaved complex)
  int i = blockIdx.x * blockDim.x + threadIdx.x;
  if (i < n1)            dout[i] = out3[i];
  else if (i < n1 + n2)  dout[i] = ((const float*)f3o)[i - n1];
}

// ---------------- host orchestration ----------------
extern "C" void kernel_launch(void* const* d_in, const int* in_sizes, int n_in,
                              void* d_out, int out_size, void* d_ws, size_t ws_size,
                              hipStream_t stream) {
  (void)in_sizes; (void)n_in; (void)out_size;
  const float*  x    = (const float*) d_in[0];
  const float2* Win  = (const float2*)d_in[1];
  const float2* Bin  = (const float2*)d_in[2];
  const float2* Wout = (const float2*)d_in[3];
  const float2* Bout = (const float2*)d_in[4];
  const float2* wq   = (const float2*)d_in[5];
  const float2* wk   = (const float2*)d_in[6];
  const float2* wv   = (const float2*)d_in[7];
  const float2* wo   = (const float2*)d_in[8];
  const float*  rmsA = (const float*) d_in[9];
  const float2* wff  = (const float2*)d_in[10];
  const float*  rmsF = (const float*) d_in[11];

  float* ws = (float*)d_ws;
  size_t off = 0;
  auto alloc = [&](size_t nf) { float* p = ws + off; off += nf; return p; };
  float*  Dre  = alloc(4096);
  float*  Dim  = alloc(4096);
  float2* f3   = (float2*)alloc(2 * (size_t)kB * kCin * kHW);
  const size_t CF = (size_t)kB * kC * kHW;     // complex elements
  float2* bufF = (float2*)alloc(2 * CF);
  float2* bufQ = (float2*)alloc(2 * CF);
  float2* bufK = (float2*)alloc(2 * CF);
  float2* bufV = (float2*)alloc(2 * CF);
  float2* bufO = (float2*)alloc(2 * CF);
  float2* bufT = (float2*)alloc(2 * CF);
  const size_t XR = (size_t)kB * kC * kH * kW;
  float*  X0   = alloc(XR);
  float*  Xn   = alloc(XR);
  float*  Xt   = alloc(XR);
  float*  Xu   = alloc(XR);
  float2* f3o  = (float2*)alloc(2 * (size_t)kB * kCin * kHW);
  float*  out3 = alloc((size_t)kB * kCin * kH * kW);
  if (off * sizeof(float) > ws_size) return;   // workspace guard

  const dim3 T(256);
  const int gGemm = kB * (kC / 16) * ((kHW / 16 + 7) / 8);   // 2*8*17 = 272
  const int gAttn = kB * (kHW / 16);                         // 264
  const int gRms  = (kB * kH * kW + 255) / 256;
  const int gWin  = (kB * kC * kHW + 255) / 256;
  const int gWou  = (kB * kCin * kHW + 255) / 256;
  const int gEle  = ((int)XR + 255) / 256;
  const int gPack = (kB * kCin * kH * kW + 2 * kB * kCin * kHW + 255) / 256;

  dft_init_kernel<<<dim3(16), T, 0, stream>>>(Dre, Dim);

  // stem: rfft2(x) -> W_in proj (+bias) -> irfft2
  rfft2_kernel<<<dim3(kB * kCin), T, 0, stream>>>(x, f3, Dre, Dim);
  win_proj_kernel<<<dim3(gWin), T, 0, stream>>>(f3, Win, Bin, bufF);
  irfft2_kernel<<<dim3(kB * kC), T, 0, stream>>>(bufF, X0, Dre, Dim);

  for (int blk = 0; blk < 2; ++blk) {
    const float2* wqb = wq  + (size_t)blk * kC * kC;
    const float2* wkb = wk  + (size_t)blk * kC * kC;
    const float2* wvb = wv  + (size_t)blk * kC * kC;
    const float2* wob = wo  + (size_t)blk * kC * kC;
    const float2* wfb = wff + (size_t)blk * kC * kC;
    // ---- attention block ----
    rmsnorm_kernel<<<dim3(gRms), T, 0, stream>>>(X0, rmsA + blk * kC, Xn);
    rfft2_kernel<<<dim3(kB * kC), T, 0, stream>>>(Xn, bufF, Dre, Dim);
    cgemm128_kernel<<<dim3(gGemm), T, 0, stream>>>(bufF, wqb, nullptr, bufQ);
    cgemm128_kernel<<<dim3(gGemm), T, 0, stream>>>(bufF, wkb, nullptr, bufK);
    cgemm128_kernel<<<dim3(gGemm), T, 0, stream>>>(bufF, wvb, nullptr, bufV);
    attn_kernel<<<dim3(gAttn), T, 0, stream>>>(bufQ, bufK, bufV, bufO);
    cgemm128_kernel<<<dim3(gGemm), T, 0, stream>>>(bufO, wob, bufF, bufT); // +xf residual
    irfft2_kernel<<<dim3(kB * kC), T, 0, stream>>>(bufT, Xt, Dre, Dim);
    // ---- feed-forward block ----
    rmsnorm_kernel<<<dim3(gRms), T, 0, stream>>>(Xt, rmsF + blk * kC, Xn);
    rfft2_kernel<<<dim3(kB * kC), T, 0, stream>>>(Xn, bufF, Dre, Dim);
    cgemm128_kernel<<<dim3(gGemm), T, 0, stream>>>(bufF, wfb, nullptr, bufT);
    irfft2_kernel<<<dim3(kB * kC), T, 0, stream>>>(bufT, Xu, Dre, Dim);
    silu_add_kernel<<<dim3(gEle), T, 0, stream>>>(Xu, Xt, X0, (int)XR);
  }

  // head: rfft2 -> W_out proj (+bias) -> irfft2 ; pack (out, xf)
  rfft2_kernel<<<dim3(kB * kC), T, 0, stream>>>(X0, bufF, Dre, Dim);
  wout_proj_kernel<<<dim3(gWou), T, 0, stream>>>(bufF, Wout, Bout, f3o);
  irfft2_kernel<<<dim3(kB * kCin), T, 0, stream>>>(f3o, out3, Dre, Dim);
  pack_kernel<<<dim3(gPack), T, 0, stream>>>(out3, f3o, (float*)d_out);
}
